// MatrixObservationModel_35828617183943
// MI455X (gfx1250) — matrix operand order repaired and checked by reading, not compiled, not run on hardware
//
#include <hip/hip_runtime.h>
#include <hip/hip_bf16.h>

typedef __attribute__((ext_vector_type(16))) __bf16 v16bf;
typedef __attribute__((ext_vector_type(8)))  __bf16 v8bf;
typedef __attribute__((ext_vector_type(8)))  float  v8f;
typedef __attribute__((ext_vector_type(4)))  unsigned int u32x4;
typedef __attribute__((ext_vector_type(8)))  int i32x8;
typedef __attribute__((ext_vector_type(4)))  int i32x4;

#define D_DIM 2496
#define B_DIM 4096
#define S_DIM 2048
#define KT    64
#define NKT   (D_DIM / KT)      // 39
#define LROW  72                // padded LDS row stride in bf16 elems (144 B = 36 dwords)
#define TILEM 128               // block tile rows (batch)
#define TILEN 256               // block tile cols (states)
#define A_SH  (TILEM * LROW)    // shorts per A LDS tile (18432 B)
#define B_SH  (TILEN * LROW)    // shorts per B LDS tile (36864 B)

// f32 -> bf16 RNE, pack pair
__device__ __forceinline__ unsigned int pk_bf16(float a, float b) {
  unsigned int ua = __float_as_uint(a), ub = __float_as_uint(b);
  ua = (ua + 0x7FFFu + ((ua >> 16) & 1u)) >> 16;
  ub = (ub + 0x7FFFu + ((ub >> 16) & 1u)) >> 16;
  return (ua & 0xFFFFu) | (ub << 16);
}
__device__ __forceinline__ uint2 pack4(float4 f) {
  uint2 r; r.x = pk_bf16(f.x, f.y); r.y = pk_bf16(f.z, f.w); return r;
}

// ---------------------------------------------------------------------------
// Fused f32->bf16 convert + row squared-norm. One 256-thread block per row.
// ---------------------------------------------------------------------------
__global__ __launch_bounds__(256) void cvt_norm_kernel(
    const float* __restrict__ in, unsigned short* __restrict__ obf,
    float* __restrict__ onorm) {
  const int r = blockIdx.x;
  const float4* p = (const float4*)(in + (size_t)r * D_DIM);
  uint2* q = (uint2*)(obf + (size_t)r * D_DIM);
  float s = 0.0f;
  for (int i = threadIdx.x; i < D_DIM / 4; i += 256) {
    float4 v = p[i];
    s += v.x * v.x + v.y * v.y + v.z * v.z + v.w * v.w;
    q[i] = pack4(v);
  }
  __shared__ float red[256];
  red[threadIdx.x] = s;
  __syncthreads();
  for (int off = 128; off > 0; off >>= 1) {
    if (threadIdx.x < off) red[threadIdx.x] += red[threadIdx.x + off];
    __syncthreads();
  }
  if (threadIdx.x == 0) onorm[r] = red[0];
}

// Plain row norm (fallback path)
__global__ __launch_bounds__(256) void row_sqnorm_kernel(
    const float* __restrict__ in, float* __restrict__ out) {
  const int r = blockIdx.x;
  const float4* p = (const float4*)(in + (size_t)r * D_DIM);
  float s = 0.0f;
  for (int i = threadIdx.x; i < D_DIM / 4; i += 256) {
    float4 v = p[i];
    s += v.x * v.x + v.y * v.y + v.z * v.z + v.w * v.w;
  }
  __shared__ float red[256];
  red[threadIdx.x] = s;
  __syncthreads();
  for (int off = 128; off > 0; off >>= 1) {
    if (threadIdx.x < off) red[threadIdx.x] += red[threadIdx.x + off];
    __syncthreads();
  }
  if (threadIdx.x == 0) out[r] = red[0];
}

// ---------------------------------------------------------------------------
// TDM: load a [rows x KT] bf16 tile (row-major, row stride D_DIM elems) into
// LDS with 16B padding per 128B row (LDS row stride 144B -> bank-conflict-free)
// ---------------------------------------------------------------------------
__device__ __forceinline__ void tdm_load_tile2d(unsigned lds_byte_off,
                                                const unsigned short* gsrc,
                                                unsigned tensor_rows,
                                                unsigned tile_rows) {
  unsigned long long ga = (unsigned long long)(uintptr_t)gsrc;
  u32x4 g0;
  g0[0] = 1u;                                   // count=1, user mode
  g0[1] = lds_byte_off;                         // lds_addr
  g0[2] = (unsigned)(ga & 0xFFFFFFFFu);         // global_addr[31:0]
  g0[3] = (unsigned)((ga >> 32) & 0x01FFFFFFu)  // global_addr[56:32]
          | 0x80000000u;                        // type=2 (bits 127:126 = 2)
  i32x8 g1;
  // data_size=1(2B), pad_enable=1, pad_interval=4(32 dw), pad_amount=3(4 dw)
  g1[0] = (1 << 16) | (1 << 20) | (4 << 22) | (3 << 25);
  g1[1] = (int)((D_DIM & 0xFFFFu) << 16);                    // tensor_dim0 lo16
  g1[2] = (int)(((D_DIM >> 16) & 0xFFFFu) |
                ((tensor_rows & 0xFFFFu) << 16));            // dim0 hi | dim1 lo
  g1[3] = (int)(((tensor_rows >> 16) & 0xFFFFu) |
                ((unsigned)KT << 16));                       // dim1 hi | tile_dim0
  g1[4] = (int)tile_rows;                                    // tile_dim1, tile_dim2=0
  g1[5] = D_DIM;                                             // tensor_dim0_stride lo32
  g1[6] = 0;
  g1[7] = 0;
  i32x4 gz = {0, 0, 0, 0};
#if __clang_major__ >= 23
  i32x8 gz8 = {0, 0, 0, 0, 0, 0, 0, 0};
  __builtin_amdgcn_tensor_load_to_lds(g0, g1, gz, gz, gz8, 0);
#else
  __builtin_amdgcn_tensor_load_to_lds(g0, g1, gz, gz, 0);
#endif
}

// ---------------------------------------------------------------------------
// Main GEMM, bf16 source: 128x256 tile/block, 64x64 per wave (4x4 WMMA),
// TDM-staged double-buffered LDS
// ---------------------------------------------------------------------------
__global__ __launch_bounds__(256) void emission_gemm_bf16_kernel(
    const unsigned short* __restrict__ Xbf,  // [B_DIM, D_DIM] bf16
    const unsigned short* __restrict__ Mbf,  // [S_DIM, D_DIM] bf16
    const float* __restrict__ xsq, const float* __restrict__ msq,
    float* __restrict__ out) {
  __shared__ __align__(16) unsigned short ldsA[2][A_SH];
  __shared__ __align__(16) unsigned short ldsB[2][B_SH];
  __shared__ float s_xsq[TILEM];
  __shared__ float s_msq[TILEN];

  const int t        = threadIdx.x;
  const int lane     = t & 31;
  const int w        = t >> 5;       // wave 0..7
  const int tileN    = blockIdx.x * TILEN;
  const int tileM    = blockIdx.y * TILEM;
  const int lane15   = lane & 15;
  const int laneHalf = lane >> 4;

  s_msq[t] = msq[tileN + t];
  if (t < TILEM) s_xsq[t] = xsq[tileM + t];

  const unsigned ldsAoff = (unsigned)(uintptr_t)(&ldsA[0][0]);
  const unsigned ldsBoff = (unsigned)(uintptr_t)(&ldsB[0][0]);
  const unsigned short* gA = Xbf + (size_t)tileM * D_DIM;
  const unsigned short* gB = Mbf + (size_t)tileN * D_DIM;

  v8f acc[4][4] = {};

  int buf = 0;
  if (w == 0) {
    tdm_load_tile2d(ldsAoff, gA, B_DIM, TILEM);
    tdm_load_tile2d(ldsBoff, gB, S_DIM, TILEN);
    __builtin_amdgcn_s_wait_tensorcnt(0);
  }
  __syncthreads();

  const int aRow0 = (w >> 2) * 64;   // 0 or 64
  const int bCol0 = (w & 3) * 64;    // 0..192

  for (int kt = 0; kt < NKT; ++kt) {
    if (kt + 1 < NKT && w == 0) {
      const int nb = buf ^ 1;
      const size_t koff = (size_t)(kt + 1) * KT;
      tdm_load_tile2d(ldsAoff + nb * (A_SH * 2), gA + koff, B_DIM, TILEM);
      tdm_load_tile2d(ldsBoff + nb * (B_SH * 2), gB + koff, S_DIM, TILEN);
    }

#pragma unroll
    for (int ks = 0; ks < 2; ++ks) {
      const int ko = ks * 32;
      v16bf afr[4];
#pragma unroll
      for (int i = 0; i < 4; ++i) {
        const unsigned short* p =
            &ldsA[buf][(aRow0 + i * 16 + lane15) * LROW + ko + laneHalf * 16];
        union { v16bf v; v8bf h[2]; } u;
        u.h[0] = *(const v8bf*)(p);
        u.h[1] = *(const v8bf*)(p + 8);
        afr[i] = u.v;
      }
      v16bf bfr[4];
#pragma unroll
      for (int j = 0; j < 4; ++j) {
        const unsigned short* p =
            &ldsB[buf][(bCol0 + j * 16 + lane15) * LROW + ko + laneHalf * 16];
        union { v16bf v; v8bf h[2]; } u;
        u.h[0] = *(const v8bf*)(p);
        u.h[1] = *(const v8bf*)(p + 8);
        bfr[j] = u.v;
      }
#pragma unroll
      for (int i = 0; i < 4; ++i)
#pragma unroll
        for (int j = 0; j < 4; ++j)
          acc[i][j] = __builtin_amdgcn_wmma_f32_16x16x32_bf16(
              false, afr[i], false, bfr[j], (short)0, acc[i][j], false, false);
    }

    if (kt + 1 < NKT) {
      if (w == 0) __builtin_amdgcn_s_wait_tensorcnt(0);
      __syncthreads();
      buf ^= 1;
    }
  }

  const float inv = -1.0f / 500.0f;
#pragma unroll
  for (int i = 0; i < 4; ++i) {
#pragma unroll
    for (int j = 0; j < 4; ++j) {
      const int colL = bCol0 + j * 16 + lane15;
      const int col  = tileN + colL;
      const float mq = s_msq[colL];
#pragma unroll
      for (int v = 0; v < 8; ++v) {
        const int rowL = aRow0 + i * 16 + laneHalf * 8 + v;
        const int row  = tileM + rowL;
        out[(size_t)row * S_DIM + col] = (s_xsq[rowL] + mq - 2.0f * acc[i][j][v]) * inv;
      }
    }
  }
}

// ---------------------------------------------------------------------------
// Fallback GEMM (f32 source, in-loop conversion) if workspace is too small
// ---------------------------------------------------------------------------
__global__ __launch_bounds__(256) void emission_gemm_f32_kernel(
    const float* __restrict__ X, const float* __restrict__ Mm,
    const float* __restrict__ xsq, const float* __restrict__ msq,
    float* __restrict__ out) {
  __shared__ __align__(32) unsigned short ldsA[2][128 * 32];
  __shared__ __align__(32) unsigned short ldsB[2][128 * 32];
  __shared__ float s_xsq[128];
  __shared__ float s_msq[128];

  const int t = threadIdx.x, lane = t & 31, w = t >> 5;
  const int tileN = blockIdx.x * 128, tileM = blockIdx.y * 128;
  const int lane15 = lane & 15, laneHalf = lane >> 4;
  const int NK32 = D_DIM / 32;

  if (t < 128) s_xsq[t] = xsq[tileM + t];
  else         s_msq[t - 128] = msq[tileN + (t - 128)];

  v8f acc[2][4] = {};
  float4 ra[4], rb[4];
#pragma unroll
  for (int i = 0; i < 4; ++i) {
    int idx = t + i * 256, row = idx >> 3, kq = idx & 7;
    ra[i] = *(const float4*)(X  + (size_t)(tileM + row) * D_DIM + kq * 4);
    rb[i] = *(const float4*)(Mm + (size_t)(tileN + row) * D_DIM + kq * 4);
  }
  int buf = 0;
#pragma unroll
  for (int i = 0; i < 4; ++i) {
    int idx = t + i * 256, row = idx >> 3, kq = idx & 7;
    *(uint2*)&ldsA[buf][row * 32 + kq * 4] = pack4(ra[i]);
    *(uint2*)&ldsB[buf][row * 32 + kq * 4] = pack4(rb[i]);
  }
  __syncthreads();

  const int aRow0 = (w >> 1) * 32, bCol0 = (w & 1) * 64;
  for (int kt = 0; kt < NK32; ++kt) {
    if (kt + 1 < NK32) {
      const int kb = (kt + 1) * 32;
#pragma unroll
      for (int i = 0; i < 4; ++i) {
        int idx = t + i * 256, row = idx >> 3, kq = idx & 7;
        ra[i] = *(const float4*)(X  + (size_t)(tileM + row) * D_DIM + kb + kq * 4);
        rb[i] = *(const float4*)(Mm + (size_t)(tileN + row) * D_DIM + kb + kq * 4);
      }
    }
    v16bf afr[2];
#pragma unroll
    for (int i = 0; i < 2; ++i) {
      const unsigned short* p = &ldsA[buf][(aRow0 + i * 16 + lane15) * 32 + laneHalf * 16];
      union { v16bf v; v8bf h[2]; } u;
      u.h[0] = *(const v8bf*)(p);
      u.h[1] = *(const v8bf*)(p + 8);
      afr[i] = u.v;
    }
    v16bf bfr[4];
#pragma unroll
    for (int j = 0; j < 4; ++j)
      bfr[j] = *(const v16bf*)(&ldsB[buf][(bCol0 + j * 16 + lane15) * 32 + laneHalf * 16]);
#pragma unroll
    for (int i = 0; i < 2; ++i)
#pragma unroll
      for (int j = 0; j < 4; ++j)
        acc[i][j] = __builtin_amdgcn_wmma_f32_16x16x32_bf16(
            false, afr[i], false, bfr[j], (short)0, acc[i][j], false, false);
    if (kt + 1 < NK32) {
      const int nb = buf ^ 1;
#pragma unroll
      for (int i = 0; i < 4; ++i) {
        int idx = t + i * 256, row = idx >> 3, kq = idx & 7;
        *(uint2*)&ldsA[nb][row * 32 + kq * 4] = pack4(ra[i]);
        *(uint2*)&ldsB[nb][row * 32 + kq * 4] = pack4(rb[i]);
      }
      __syncthreads();
      buf = nb;
    }
  }
  const float inv = -1.0f / 500.0f;
#pragma unroll
  for (int i = 0; i < 2; ++i)
#pragma unroll
    for (int j = 0; j < 4; ++j) {
      const int colL = bCol0 + j * 16 + lane15, col = tileN + colL;
      const float mq = s_msq[colL];
#pragma unroll
      for (int v = 0; v < 8; ++v) {
        const int rowL = aRow0 + i * 16 + laneHalf * 8 + v;
        out[(size_t)(tileM + rowL) * S_DIM + col] =
            (s_xsq[rowL] + mq - 2.0f * acc[i][j][v]) * inv;
      }
    }
}

// ---------------------------------------------------------------------------
extern "C" void kernel_launch(void* const* d_in, const int* in_sizes, int n_in,
                              void* d_out, int out_size, void* d_ws, size_t ws_size,
                              hipStream_t stream) {
  const float* X  = (const float*)d_in[0];
  const float* Mm = (const float*)d_in[1];
  float* out = (float*)d_out;

  const size_t nX = (size_t)B_DIM * D_DIM;
  const size_t nM = (size_t)S_DIM * D_DIM;
  const size_t need = (nX + nM) * sizeof(unsigned short) +
                      (B_DIM + S_DIM) * sizeof(float);

  if (ws_size >= need) {
    unsigned short* Xbf = (unsigned short*)d_ws;
    unsigned short* Mbf = Xbf + nX;
    float* xsq = (float*)(Mbf + nM);
    float* msq = xsq + B_DIM;
    cvt_norm_kernel<<<B_DIM, 256, 0, stream>>>(X, Xbf, xsq);
    cvt_norm_kernel<<<S_DIM, 256, 0, stream>>>(Mm, Mbf, msq);
    dim3 grid(S_DIM / TILEN, B_DIM / TILEM);  // 8 x 32
    emission_gemm_bf16_kernel<<<grid, 256, 0, stream>>>(Xbf, Mbf, xsq, msq, out);
  } else {
    float* xsq = (float*)d_ws;
    float* msq = xsq + B_DIM;
    row_sqnorm_kernel<<<B_DIM, 256, 0, stream>>>(X, xsq);
    row_sqnorm_kernel<<<S_DIM, 256, 0, stream>>>(Mm, msq);
    dim3 grid(S_DIM / 128, B_DIM / 128);      // 16 x 32
    emission_gemm_f32_kernel<<<grid, 256, 0, stream>>>(X, Mm, xsq, msq, out);
  }
}
